// PAConv_54614804136179
// MI455X (gfx1250) — matrix operand order repaired and checked by reading, not compiled, not run on hardware
//
#include <hip/hip_runtime.h>
#include <hip/hip_bf16.h>

typedef __attribute__((ext_vector_type(16))) __bf16 v16bf;
typedef __attribute__((ext_vector_type(8)))  __bf16 v8bf;
typedef __attribute__((ext_vector_type(8)))  float  v8f;
typedef __attribute__((ext_vector_type(4)))  unsigned int u32x4;
typedef __attribute__((ext_vector_type(4)))  int          i32x4;
typedef __attribute__((ext_vector_type(8)))  int          i32x8;

#define EPS_F 1e-5f

// ---- problem constants ----
constexpr int B_    = 8;
constexpr int CIN   = 64;
constexpr int N_    = 1024;
constexpr int KNB   = 32;            // neighbors
constexpr int COUT  = 64;
constexpr int M_    = 8;
constexpr int PPB   = N_ * KNB;      // 32768 points per batch
constexpr int TP    = 128;           // points per tile
constexpr int TILES = (B_ * PPB) / TP;   // 2048
constexpr int TPB   = PPB / TP;          // 256 tiles per batch
constexpr int KDIM  = 2 * CIN * M_;      // 1024  (GEMM reduction dim: K = c*8+m)
constexpr int NCHUNK = KDIM / 32;        // 32 chunks of 32

// ---- LDS layout (byte offsets into dynamic smem) ----
constexpr int W_ROWU  = 72;                              // sizing only; W rows are o x (KDIM+8) bf16
constexpr int W_OFF   = 0;                               // 1024 * 144B = 147456
constexpr int X_OFF   = KDIM * W_ROWU * 2;               // 147456
constexpr int X_ROWB  = 80;                              // 32 bf16 (64B) + 16B pad
constexpr int X_BUFB  = TP * X_ROWB;                     // 10240 per buffer (x2)
constexpr int FL_OFF  = X_OFF + 2 * X_BUFB;              // 167936 : 64 x 128 f32 raw features
constexpr int S_OFF   = FL_OFF + CIN * TP * 4;           // 200704 : 128 x 8 f32 scores
constexpr int WM_OFF  = S_OFF + TP * M_ * 4;             // 204800 : MLP/BN params (<=1024 f32)
constexpr int OUT_OFF = WM_OFF + 4096;                   // 208896 : 64 x 132 f32 out staging
constexpr int OUT_ROW = TP + 4;                          // 132 floats (bank derotate)
constexpr int LDS_BYTES = OUT_OFF + COUT * OUT_ROW * 4;  // 242688 < 320KB

__device__ __forceinline__ unsigned short f2bf(float f) {
  union { float f; unsigned u; } v; v.f = f;
  unsigned u = v.u;
  u += 0x7fffu + ((u >> 16) & 1u);   // round-to-nearest-even
  return (unsigned short)(u >> 16);
}

__device__ __forceinline__ v16bf cat16(v8bf lo, v8bf hi) {
  return __builtin_shufflevector(lo, hi, 0,1,2,3,4,5,6,7,8,9,10,11,12,13,14,15);
}

// TDM: one 2D tile (tile_dim0=128 f32 elems, tile_dim1=64 lines, line stride 32768 elems)
// from global `ga` into LDS byte-address `lds_addr`. D# packing per CDNA5 ISA.
__device__ __forceinline__ void tdm_load_feature_slab(unsigned long long ga, unsigned lds_addr) {
  u32x4 g0;
  g0[0] = 1u;                                   // count=1, is_restore=0, gather off
  g0[1] = lds_addr;                             // lds_addr [63:32]
  g0[2] = (unsigned)ga;                         // global_addr [95:64]
  g0[3] = ((unsigned)(ga >> 32) & 0x01FFFFFFu)  // global_addr [120:96]
          | (2u << 30);                         // type=2 ("image") [127:126]
  i32x8 g1;
  g1[0] = 0x00020000;                           // wg_mask=0, data_size=2 (4B), no flags
  g1[1] = (int)(32768u << 16);                  // atomic_addr=0 | tensor_dim0[15:0]=32768 -> 0x80000000
  g1[2] = (int)(64u << 16);                     // tensor_dim0[31:16]=0 | tensor_dim1[15:0]=64
  g1[3] = (int)(128u << 16);                    // tensor_dim1[31:16]=0 | tile_dim0=128
  g1[4] = 64;                                   // tile_dim1=64 | tile_dim2=0
  g1[5] = 32768;                                // tensor_dim0_stride[31:0]
  g1[6] = 0;                                    // stride hi | dim1_stride lo
  g1[7] = 0;
  i32x4 z4 = {0, 0, 0, 0};
#if defined(__clang_major__) && (__clang_major__ >= 23)
  i32x8 z8 = {0, 0, 0, 0, 0, 0, 0, 0};
  __builtin_amdgcn_tensor_load_to_lds(g0, g1, z4, z4, z8, 0);
#else
  __builtin_amdgcn_tensor_load_to_lds(g0, g1, z4, z4, 0);
#endif
}

// Build one 128x32 bf16 X slab: X[p][kk] = f2[p,c]*s[p,m], kk=(c-cbase)*8+m
__device__ __forceinline__ void build_x_chunk(int tt, char* xbuf, const float* FL,
                                              const float* SL, int tid) {
  int p = tid >> 1, hf = tid & 1;
  int cA = tt * 4 + hf * 2;
  int p0 = p & ~31;                 // center index (k==0 of this n)
  float sv[8];
  #pragma unroll
  for (int m = 0; m < 8; ++m) sv[m] = SL[p * 8 + m];
  int crA = cA & 63;       float cmA = (cA < CIN) ? 1.f : 0.f;
  int crB = (cA + 1) & 63; float cmB = ((cA + 1) < CIN) ? 1.f : 0.f;
  float fA = FL[crA * TP + p] - cmA * FL[crA * TP + p0];
  float fB = FL[crB * TP + p] - cmB * FL[crB * TP + p0];
  unsigned u[8];
  #pragma unroll
  for (int j = 0; j < 4; ++j)
    u[j] = (unsigned)f2bf(fA * sv[2 * j]) | ((unsigned)f2bf(fA * sv[2 * j + 1]) << 16);
  #pragma unroll
  for (int j = 0; j < 4; ++j)
    u[4 + j] = (unsigned)f2bf(fB * sv[2 * j]) | ((unsigned)f2bf(fB * sv[2 * j + 1]) << 16);
  char* xr = xbuf + p * X_ROWB + hf * 32;
  uint4 t0; t0.x = u[0]; t0.y = u[1]; t0.z = u[2]; t0.w = u[3];
  uint4 t1; t1.x = u[4]; t1.y = u[5]; t1.z = u[6]; t1.w = u[7];
  *(uint4*)(xr)      = t0;
  *(uint4*)(xr + 16) = t1;
}

__global__ void paconv_wmma_kernel(
    const float* __restrict__ feat, const float* __restrict__ xyz,
    const float* __restrict__ w1, const float* __restrict__ g1, const float* __restrict__ be1,
    const float* __restrict__ w2, const float* __restrict__ g2, const float* __restrict__ be2,
    const float* __restrict__ w3, const float* __restrict__ g3, const float* __restrict__ be3,
    const float* __restrict__ w4, const float* __restrict__ b4,
    const float* __restrict__ wb,
    const float* __restrict__ bng, const float* __restrict__ bnb,
    float* __restrict__ outp)
{
  extern __shared__ char smem[];
  unsigned short* Wus = (unsigned short*)(smem + W_OFF);
  char*  Xb = smem + X_OFF;
  float* FL = (float*)(smem + FL_OFF);
  float* SL = (float*)(smem + S_OFF);
  float* WM = (float*)(smem + WM_OFF);
  float* OS = (float*)(smem + OUT_OFF);

  const int tid  = threadIdx.x;
  const int lane = tid & 31;
  const int wave = tid >> 5;            // 8 waves -> 8 row bands of 16 points
  const float rs = rsqrtf(1.0f + EPS_F);
  const unsigned fl_lds_addr = (unsigned)(unsigned long long)(uintptr_t)(void*)FL;

  // ---- stage weight bank fp32 -> bf16 into LDS (Wt[o][K], row stride KDIM+8, K = c*8+m == flat row) ----
  for (int i = tid; i < KDIM * COUT; i += 256) {
    int k = i >> 6, o = i & 63;
    Wus[o * (KDIM + 8) + k] = f2bf(wb[i]);
  }
  // ---- stage ScoreNet + BN params (fold bn scale = g * rsqrt(1+eps)) ----
  for (int i = tid; i < 984; i += 256) {
    float v;
    if      (i < 112) v = w1[i];
    else if (i < 128) v = g1[i - 112] * rs;
    else if (i < 144) v = be1[i - 128];
    else if (i < 400) v = w2[i - 144];
    else if (i < 416) v = g2[i - 400] * rs;
    else if (i < 432) v = be2[i - 416];
    else if (i < 688) v = w3[i - 432];
    else if (i < 704) v = g3[i - 688] * rs;
    else if (i < 720) v = be3[i - 704];
    else if (i < 848) v = w4[i - 720];
    else if (i < 856) v = b4[i - 848];
    else if (i < 920) v = bng[i - 856] * rs;
    else              v = bnb[i - 920];
    WM[i] = v;
  }
  __syncthreads();

  // ---- persistent loop over point tiles ----
  for (int tile = blockIdx.x; tile < TILES; tile += gridDim.x) {
    const int b   = tile / TPB;
    const int off = (tile - b * TPB) * TP;     // flat (n*32+k) base, multiple of 128
    const float* fb = feat + (size_t)b * CIN * PPB + off;

    // TDM: wave 0 pulls the 64x512B feature slab straight into LDS
    if (wave == 0) {
      tdm_load_feature_slab((unsigned long long)(uintptr_t)fb, fl_lds_addr);
      __builtin_amdgcn_s_wait_tensorcnt(0);
    }

    // prefetch next tile's slab into L2 while this tile computes
    {
      int nt = tile + gridDim.x;
      if (nt < TILES) {
        int b2 = nt / TPB;
        const float* fb2 = feat + (size_t)b2 * CIN * PPB + (nt - b2 * TPB) * TP;
        int c = tid & 63, q = (tid >> 6) << 5;   // 4 x 128B per 512B row
        __builtin_prefetch(fb2 + (size_t)c * PPB + q, 0, 0);
      }
    }

    // ---- ScoreNet MLP + softmax, one thread per point ----
    if (tid < TP) {
      int flat = off + tid;
      int n = flat >> 5, k = flat & 31;
      const float* xb = xyz + (size_t)b * 3 * PPB;
      int nb = n << 5;
      float cx = xb[nb],           cy = xb[PPB + nb],     cz = xb[2 * PPB + nb];
      float px = xb[nb + k],       py = xb[PPB + nb + k], pz = xb[2 * PPB + nb + k];
      float dx = px - cx, dy = py - cy, dz = pz - cz;
      float dist = sqrtf(dx * dx + dy * dy + dz * dz);
      float in7[7] = {cx, cy, cz, dx, dy, dz, dist};
      float h1[16], h2[16], h3[16];
      #pragma unroll
      for (int i = 0; i < 16; ++i) {
        float a = 0.f;
        #pragma unroll
        for (int j = 0; j < 7; ++j) a += WM[i * 7 + j] * in7[j];
        h1[i] = fmaxf(a * WM[112 + i] + WM[128 + i], 0.f);
      }
      #pragma unroll
      for (int i = 0; i < 16; ++i) {
        float a = 0.f;
        #pragma unroll
        for (int j = 0; j < 16; ++j) a += WM[144 + i * 16 + j] * h1[j];
        h2[i] = fmaxf(a * WM[400 + i] + WM[416 + i], 0.f);
      }
      #pragma unroll
      for (int i = 0; i < 16; ++i) {
        float a = 0.f;
        #pragma unroll
        for (int j = 0; j < 16; ++j) a += WM[432 + i * 16 + j] * h2[j];
        h3[i] = fmaxf(a * WM[688 + i] + WM[704 + i], 0.f);
      }
      float lg[8]; float mx = -1e30f;
      #pragma unroll
      for (int m = 0; m < 8; ++m) {
        float a = WM[848 + m];
        #pragma unroll
        for (int j = 0; j < 16; ++j) a += WM[720 + m * 16 + j] * h3[j];
        lg[m] = a; mx = fmaxf(mx, a);
      }
      float sm = 0.f;
      #pragma unroll
      for (int m = 0; m < 8; ++m) { float e = __expf(lg[m] - mx); lg[m] = e; sm += e; }
      float inv = 1.f / sm;
      #pragma unroll
      for (int m = 0; m < 8; ++m) SL[tid * 8 + m] = lg[m] * inv;
    }
    __syncthreads();   // FL (TDM) + SL visible to all waves

    v8f acc0 = {}, acc1 = {}, acc2 = {}, acc3 = {};

    // ---- K loop: 32 chunks of K=32, double-buffered X, 1 barrier/chunk ----
    build_x_chunk(0, Xb, FL, SL, tid);
    __syncthreads();
    for (int tt = 0; tt < NCHUNK; ++tt) {
      char* cur = Xb + (tt & 1) * X_BUFB;
      char* nxt = Xb + ((tt + 1) & 1) * X_BUFB;
      if (tt + 1 < NCHUNK) build_x_chunk(tt + 1, nxt, FL, SL, tid);

      // WMMA: A = 16x32 band of X, B = 32x16 slice of W, f32 accumulate
      {
        int r = lane & 15, hf2 = lane >> 4;
        const char* ar = cur + (wave * 16 + r) * X_ROWB + hf2 * 16;
        v16bf a = cat16(*(const v8bf*)(ar), *(const v8bf*)(ar + 32));
        const char* wr = (const char*)Wus + (size_t)r * ((KDIM + 8) * 2) + (tt * 32 + hf2 * 8) * 2;
        v16bf b0 = cat16(*(const v8bf*)(wr), *(const v8bf*)(wr + 32));
        v16bf b1 = cat16(*(const v8bf*)(wr + 16 * (KDIM + 8) * 2), *(const v8bf*)(wr + 16 * (KDIM + 8) * 2 + 32));
        v16bf b2 = cat16(*(const v8bf*)(wr + 32 * (KDIM + 8) * 2), *(const v8bf*)(wr + 32 * (KDIM + 8) * 2 + 32));
        v16bf b3 = cat16(*(const v8bf*)(wr + 48 * (KDIM + 8) * 2), *(const v8bf*)(wr + 48 * (KDIM + 8) * 2 + 32));
        acc0 = __builtin_amdgcn_wmma_f32_16x16x32_bf16(false, a, false, b0, (short)0, acc0, false, false);
        acc1 = __builtin_amdgcn_wmma_f32_16x16x32_bf16(false, a, false, b1, (short)0, acc1, false, false);
        acc2 = __builtin_amdgcn_wmma_f32_16x16x32_bf16(false, a, false, b2, (short)0, acc2, false, false);
        acc3 = __builtin_amdgcn_wmma_f32_16x16x32_bf16(false, a, false, b3, (short)0, acc3, false, false);
      }
      __syncthreads();
    }

    // ---- epilogue: BN + ReLU, transpose through LDS ----
    {
      int r = lane & 15, hf2 = lane >> 4;
      #pragma unroll
      for (int ct = 0; ct < 4; ++ct) {
        v8f av = (ct == 0) ? acc0 : (ct == 1) ? acc1 : (ct == 2) ? acc2 : acc3;
        int o = ct * 16 + r;
        float sc = WM[856 + o], bi = WM[920 + o];
        #pragma unroll
        for (int v = 0; v < 8; ++v) {
          int pl = wave * 16 + v + hf2 * 8;
          OS[o * OUT_ROW + pl] = fmaxf(av[v] * sc + bi, 0.f);
        }
      }
    }
    __syncthreads();

    // contiguous 512B-row stores
    {
      float* ob = outp + (size_t)b * COUT * PPB + off;
      #pragma unroll
      for (int i = 0; i < 8; ++i) {
        int j = i * 256 + tid;
        int o = j >> 5, q = (j & 31) << 2;
        float4 v = *(const float4*)(OS + o * OUT_ROW + q);
        *(float4*)(ob + (size_t)o * PPB + q) = v;
      }
    }
    __syncthreads();
  }
}

extern "C" void kernel_launch(void* const* d_in, const int* in_sizes, int n_in,
                              void* d_out, int out_size, void* d_ws, size_t ws_size,
                              hipStream_t stream) {
  const float* feat = (const float*)d_in[0];
  const float* xyz  = (const float*)d_in[1];
  const float* w1   = (const float*)d_in[2];
  const float* g1   = (const float*)d_in[3];
  const float* be1  = (const float*)d_in[4];
  const float* w2   = (const float*)d_in[5];
  const float* g2   = (const float*)d_in[6];
  const float* be2  = (const float*)d_in[7];
  const float* w3   = (const float*)d_in[8];
  const float* g3   = (const float*)d_in[9];
  const float* be3  = (const float*)d_in[10];
  const float* w4   = (const float*)d_in[11];
  const float* b4   = (const float*)d_in[12];
  const float* wb   = (const float*)d_in[13];
  const float* bng  = (const float*)d_in[14];
  const float* bnb  = (const float*)d_in[15];
  float* outp = (float*)d_out;

  paconv_wmma_kernel<<<dim3(256), dim3(256), LDS_BYTES, stream>>>(
      feat, xyz, w1, g1, be1, w2, g2, be2, w3, g3, be3, w4, b4, wb, bng, bnb, outp);
}
